// DescrptSeA_19739669692597
// MI455X (gfx1250) — compile-verified, matrix operand order checked by reading, not run on hardware
//
#include <hip/hip_runtime.h>
#include <hip/hip_bf16.h>
#include <math.h>

#define NF_    2
#define NLOC_  4096
#define NALL_  4608
#define NNEI_  138
#define NT_    2
#define MOUT_  100
#define AXIS_  16
#define SEC1_  46      // type-0 neighbors: [0,46); type-1: [46,138)
#define RPAD_  144     // 48 padded rows (type 0) + 96 padded rows (type 1)
#define BLK_   128     // 4 wave32
#define APW_   4       // atoms per workgroup (amortize weight staging)

typedef __attribute__((ext_vector_type(16))) _Float16 v16h;
typedef __attribute__((ext_vector_type(8)))  _Float16 v8h;
typedef __attribute__((ext_vector_type(8)))  float    v8f;

// Branch-free tanh: 1 - 2/(exp(2x)+1).  exp->v_exp_f32, rcp->v_rcp_f32.
// x->+inf: rcp(inf)=0 -> 1;  x->-inf: rcp(1)=1 -> -1.  No NaN for finite x.
__device__ __forceinline__ float fast_tanh(float x) {
  float e = __expf(2.0f * x);
  float r = __builtin_amdgcn_rcpf(e + 1.0f);
  return fmaf(-2.0f, r, 1.0f);
}

// Assemble one wave32 WMMA 16-bit operand register set from an LDS row laid
// out row-major with 64B-aligned rows. Per CDNA5 ISA (CDNA5 ISA):
//   lane L holds row (L&15); kbase = (L<16)?0:8;
//   v16h elems [0..7]  = row[kbase + 0..7]      (VGPR0..3)
//   v16h elems [8..15] = row[kbase + 16..23]    (VGPR4..7)
__device__ __forceinline__ v16h lds_tile_row(const _Float16* __restrict__ rowp,
                                             int kbase) {
  const v8h* lo = reinterpret_cast<const v8h*>(rowp + kbase);
  const v8h* hi = reinterpret_cast<const v8h*>(rowp + kbase + 16);
  v16h v;
#pragma unroll
  for (int i = 0; i < 8; ++i) v[i] = (*lo)[i];
#pragma unroll
  for (int i = 0; i < 8; ++i) v[8 + i] = (*hi)[i];
  return v;
}

__global__ __launch_bounds__(BLK_)
void descrpt_sea_fused(const int* __restrict__ nlist,
                       const float* __restrict__ coord,
                       const int* __restrict__ atype,
                       const float* __restrict__ mean,
                       const float* __restrict__ stdv,
                       const float* __restrict__ w0,
                       const float* __restrict__ b0,
                       const float* __restrict__ w1,
                       const float* __restrict__ b1,
                       const float* __restrict__ w2,
                       const float* __restrict__ b2,
                       float* __restrict__ out) {
  // ---- LDS: 65,344 B total ----
  __shared__ __align__(16) _Float16 h0h[RPAD_][32];    //  9216 B  (A, layer1)
  __shared__ __align__(16) _Float16 h1h[RPAD_][64];    // 18432 B  (A, layer2)
  __shared__ __align__(16) _Float16 w1t[NT_][50][32];  //  6400 B  (B^T, layer1)
  __shared__ __align__(16) _Float16 w2t[NT_][100][64]; // 25600 B  (B^T, layer2)
  __shared__ __align__(16) float    rr_s[RPAD_][4];    //  2304 B
  __shared__ float    w0s[NT_][25], b0s[NT_][25];      //   400 B
  __shared__ float    b1s[NT_][50];                    //   400 B
  __shared__ float    b2s[NT_][100];                   //   800 B
  __shared__ float    xyzacc[4][112];                  //  1792 B

  const int tid  = threadIdx.x;
  const int wave = tid >> 5;
  const int lane = tid & 31;

  // ---- stage weights once per workgroup (f16, transposed, zero-padded K) ----
  for (int i = tid; i < NT_ * 25; i += BLK_) {
    int t = i / 25, k = i % 25;
    w0s[t][k] = w0[i];                    // (NT,1,25) flat == (NT,25)
    b0s[t][k] = b0[i];
  }
  for (int i = tid; i < NT_ * 50;  i += BLK_) b1s[i / 50][i % 50]   = b1[i];
  for (int i = tid; i < NT_ * 100; i += BLK_) b2s[i / 100][i % 100] = b2[i];
  for (int i = tid; i < NT_ * 50 * 32; i += BLK_) {
    int t = i / (50 * 32), r = i % (50 * 32);
    int n = r >> 5, k = r & 31;           // w1t[t][n][k] = w1[t][k][n]
    w1t[t][n][k] = (_Float16)((k < 25) ? w1[(t * 25 + k) * 50 + n] : 0.0f);
  }
  for (int i = tid; i < NT_ * 100 * 64; i += BLK_) {
    int t = i / (100 * 64), r = i % (100 * 64);
    int n = r >> 6, k = r & 63;           // w2t[t][n][k] = w2[t][k][n]
    w2t[t][n][k] = (_Float16)((k < 50) ? w2[(t * 50 + k) * 100 + n] : 0.0f);
  }
  // zero rr at segment-padding rows {46,47,140..143} (stay zero for all atoms)
  for (int idx = tid; idx < 6; idx += BLK_) {
    int row = (idx < 2) ? (46 + idx) : (138 + idx);
#pragma unroll
    for (int c = 0; c < 4; ++c) rr_s[row][c] = 0.0f;
  }
  __syncthreads();

  const int kbase = (lane < 16) ? 0 : 8;

  for (int ai = 0; ai < APW_; ++ai) {
    const int atom = blockIdx.x * APW_ + ai;      // 0 .. NF*NLOC-1
    const int f    = atom / NLOC_;
    const int iloc = atom - f * NLOC_;

    // ---- Phase A1: environment matrix rr (per neighbor); zero xyzacc ----
    for (int i = tid; i < 4 * 112; i += BLK_) xyzacc[i / 112][i % 112] = 0.0f;

    const int   tcen = atype[f * NALL_ + iloc];
    const float ccx  = coord[(f * NALL_ + iloc) * 3 + 0];
    const float ccy  = coord[(f * NALL_ + iloc) * 3 + 1];
    const float ccz  = coord[(f * NALL_ + iloc) * 3 + 2];

    for (int j = tid; j < NNEI_; j += BLK_) {
      int nid    = nlist[atom * NNEI_ + j];
      bool valid = (nid >= 0);
      int  nl    = valid ? nid : 0;
      float dx = coord[(f * NALL_ + nl) * 3 + 0] - ccx;
      float dy = coord[(f * NALL_ + nl) * 3 + 1] - ccy;
      float dz = coord[(f * NALL_ + nl) * 3 + 2] - ccz;
      float r  = sqrtf(dx * dx + dy * dy + dz * dz);
      float uu = (r - 0.5f) * (1.0f / 5.5f);
      float vv = uu * uu * uu * (-6.0f * uu * uu + 15.0f * uu - 10.0f) + 1.0f;
      float w  = (r < 0.5f) ? 1.0f : ((r >= 6.0f) ? 0.0f : vv);
      if (!valid) w = 0.0f;
      float inv = 1.0f / (r + 0.01f);
      float c1  = inv * inv * w;
      float env[4] = { inv * w, dx * c1, dy * c1, dz * c1 };

      const int row = (j < SEC1_) ? j : j + 2;   // segment-padded row index
      const float* mp = mean + (tcen * NNEI_ + j) * 4;
      const float* sp = stdv + (tcen * NNEI_ + j) * 4;
#pragma unroll
      for (int c = 0; c < 4; ++c) rr_s[row][c] = (env[c] - mp[c]) / sp[c];
    }
    __syncthreads();

    // ---- Phase A2: layer0  h0 = tanh(s*w0 + b0)  over full padded grid ----
    for (int i = tid; i < RPAD_ * 32; i += BLK_) {
      int row = i >> 5, k = i & 31;
      int t   = (row < 48) ? 0 : 1;
      float hv = 0.0f;
      if (k < 25)   // pad rows have rr_s==0 -> finite tanh(b0); harmless
        hv = fast_tanh(rr_s[row][0] * w0s[t][k] + b0s[t][k]);
      h0h[row][k] = (_Float16)hv;
    }
    __syncthreads();

    // ---- Layer1: (144 x 25) @ (25 x 50), one WMMA per 16x16 tile ----
    for (int tile = wave; tile < 9 * 4; tile += 4) {  // wave-uniform: EXEC full
      const int mt = tile >> 2;
      const int nt = tile & 3;
      const int t  = (mt < 3) ? 0 : 1;
      const int arow = mt * 16 + (lane & 15);
      const int n    = nt * 16 + (lane & 15);
      const bool keep = (n < 50);
      const int  nc   = keep ? n : 49;               // clamped, safe LDS index

      v16h a = lds_tile_row(&h0h[arow][0],   kbase);
      v16h b = lds_tile_row(&w1t[t][nc][0],  kbase);
      v8f  c = {};
      c = __builtin_amdgcn_wmma_f32_16x16x32_f16(false, a, false, b,
                                                 (short)0, c, false, false);
      const float bias    = b1s[t][nc];              // hoisted, g-invariant
      const int   skipIdx = (nc < 25) ? nc : nc - 25;
#pragma unroll
      for (int g = 0; g < 8; ++g) {
        int row = mt * 16 + g + ((lane < 16) ? 0 : 8);
        float h0skip = (float)h0h[row][skipIdx];
        float val = fast_tanh(c[g] + bias) + h0skip;
        h1h[row][n] = (_Float16)(keep ? val : 0.0f); // zeroes K-pad cols 50..63
      }
    }
    __syncthreads();

    // ---- Layer2: (144 x 50) @ (50 x 100), 2 chained WMMA per tile,
    //      fused gg epilogue + xyz += rr^T @ gg (LDS float atomics) ----
    for (int tile = wave; tile < 9 * 7; tile += 4) {  // wave-uniform: EXEC full
      const int mt = tile / 7;
      const int nt = tile % 7;
      const int t  = (mt < 3) ? 0 : 1;
      const int arow = mt * 16 + (lane & 15);
      const int n    = nt * 16 + (lane & 15);
      const bool keep = (n < MOUT_);
      const int  nc   = keep ? n : 99;               // clamped, safe LDS index

      v8f c = {};
      {
        v16h a = lds_tile_row(&h1h[arow][0],    kbase);   // K = 0..31
        v16h b = lds_tile_row(&w2t[t][nc][0],   kbase);
        c = __builtin_amdgcn_wmma_f32_16x16x32_f16(false, a, false, b,
                                                   (short)0, c, false, false);
        a = lds_tile_row(&h1h[arow][32],  kbase);         // K = 32..63
        b = lds_tile_row(&w2t[t][nc][32], kbase);
        c = __builtin_amdgcn_wmma_f32_16x16x32_f16(false, a, false, b,
                                                   (short)0, c, false, false);
      }
      const float bias    = b2s[t][nc];              // hoisted, g-invariant
      const int   skipIdx = (nc < 50) ? nc : nc - 50;
      float acc0 = 0.f, acc1 = 0.f, acc2 = 0.f, acc3 = 0.f;
#pragma unroll
      for (int g = 0; g < 8; ++g) {
        int row = mt * 16 + g + ((lane < 16) ? 0 : 8);
        float h1skip = (float)h1h[row][skipIdx];
        float gv = fast_tanh(c[g] + bias) + h1skip;  // gg[row][n]
        float4 rv = *reinterpret_cast<const float4*>(&rr_s[row][0]);
        acc0 = fmaf(gv, rv.x, acc0);
        acc1 = fmaf(gv, rv.y, acc1);
        acc2 = fmaf(gv, rv.z, acc2);
        acc3 = fmaf(gv, rv.w, acc3);
      }
      if (keep) {
        atomicAdd(&xyzacc[0][n], acc0);
        atomicAdd(&xyzacc[1][n], acc1);
        atomicAdd(&xyzacc[2][n], acc2);
        atomicAdd(&xyzacc[3][n], acc3);
      }
    }
    __syncthreads();

    // ---- Output: res[m][a] = sum_d (xyz[d][m]/NNEI)*(xyz[d][a]/NNEI) ----
    const float invn = 1.0f / (float)NNEI_;
    float* po = out + (size_t)atom * (MOUT_ * AXIS_);
    for (int i = tid; i < MOUT_ * AXIS_; i += BLK_) {
      int m = i >> 4;
      int a = i & 15;
      float s = 0.0f;
#pragma unroll
      for (int d = 0; d < 4; ++d) s += xyzacc[d][m] * xyzacc[d][a];
      po[i] = s * invn * invn;
    }
    __syncthreads();   // protect next iteration's rr_s/xyzacc writes
  }
}

extern "C" void kernel_launch(void* const* d_in, const int* in_sizes, int n_in,
                              void* d_out, int out_size, void* d_ws, size_t ws_size,
                              hipStream_t stream) {
  (void)in_sizes; (void)n_in; (void)out_size; (void)d_ws; (void)ws_size;
  const int*   nlist = (const int*)  d_in[0];
  const float* coord = (const float*)d_in[1];
  const int*   atype = (const int*)  d_in[2];
  const float* mean  = (const float*)d_in[3];
  const float* stdv  = (const float*)d_in[4];
  const float* w0    = (const float*)d_in[5];
  const float* b0    = (const float*)d_in[6];
  const float* w1    = (const float*)d_in[7];
  const float* b1    = (const float*)d_in[8];
  const float* w2    = (const float*)d_in[9];
  const float* b2    = (const float*)d_in[10];
  float* out = (float*)d_out;

  dim3 grid((NF_ * NLOC_) / APW_);
  dim3 block(BLK_);
  hipLaunchKernelGGL(descrpt_sea_fused, grid, block, 0, stream,
                     nlist, coord, atype, mean, stdv,
                     w0, b0, w1, b1, w2, b2, out);
}
